// MultiHeadCrossAttention_38001870635532
// MI455X (gfx1250) — matrix operand order repaired and checked by reading, not compiled, not run on hardware
//
#include <hip/hip_runtime.h>
#include <hip/hip_bf16.h>

// Problem constants (match reference)
#define BB   2
#define NN   2048
#define MM   4096
#define DD   256
#define HH   8
#define DKK  32
#define KNNB 16
#define BN   (BB*NN)   // 4096 query rows
#define BM   (BB*MM)   // 8192 key rows

typedef __attribute__((ext_vector_type(16))) __bf16 v16bf;
typedef __attribute__((ext_vector_type(8)))  __bf16 v8bf;
typedef __attribute__((ext_vector_type(8)))  float  v8f;

// ---------------------------------------------------------------------------
// f32 -> bf16 elementwise convert
// ---------------------------------------------------------------------------
__global__ __launch_bounds__(256)
void cvt_bf16_kernel(const float* __restrict__ in, __bf16* __restrict__ out, int n) {
    int i = blockIdx.x * 256 + threadIdx.x;
    if (i < n) out[i] = (__bf16)in[i];
}

// ---------------------------------------------------------------------------
// Transpose + convert a 256x256 weight (din,dout) -> WT[dout][din] in bf16
// ---------------------------------------------------------------------------
__global__ __launch_bounds__(256)
void transpose_cvt_kernel(const float* __restrict__ W, __bf16* __restrict__ WT) {
    int i = blockIdx.x * 256 + threadIdx.x;   // i over 65536
    int k = i >> 8;        // din
    int n = i & 255;       // dout
    WT[(size_t)n * DD + k] = (__bf16)W[i];
}

// ---------------------------------------------------------------------------
// kNN: one block per query row. Squared distances staged in LDS, then 16
// serial argmin passes (tie -> lower index, matching lax.top_k ordering).
// ---------------------------------------------------------------------------
__global__ __launch_bounds__(256)
void knn_kernel(const float* __restrict__ src, const float* __restrict__ tgt,
                int* __restrict__ knn_idx) {
    __shared__ float dist[MM];
    __shared__ float rv[256];
    __shared__ int   ri[256];
    const int row = blockIdx.x;               // 0..BN-1
    const int b   = row / NN;
    const int tid = threadIdx.x;

    const float sx = src[(size_t)row * 3 + 0];
    const float sy = src[(size_t)row * 3 + 1];
    const float sz = src[(size_t)row * 3 + 2];

    for (int m = tid; m < MM; m += 256) {
        const float* t = tgt + ((size_t)b * MM + m) * 3;
        float dx = sx - t[0], dy = sy - t[1], dz = sz - t[2];
        dist[m] = dx * dx + dy * dy + dz * dz;
    }
    __syncthreads();

    for (int k = 0; k < KNNB; ++k) {
        float best = 3.4e38f; int bi = MM;
        for (int m = tid; m < MM; m += 256) {
            float d = dist[m];
            if (d < best) { best = d; bi = m; }
        }
        rv[tid] = best; ri[tid] = bi;
        __syncthreads();
        for (int s = 128; s > 0; s >>= 1) {
            if (tid < s) {
                float ov = rv[tid + s]; int oi = ri[tid + s];
                if (ov < rv[tid] || (ov == rv[tid] && oi < ri[tid])) { rv[tid] = ov; ri[tid] = oi; }
            }
            __syncthreads();
        }
        if (tid == 0) {
            knn_idx[(size_t)row * KNNB + k] = ri[0];
            dist[ri[0]] = 3.4e38f;            // exclude from later passes
        }
        __syncthreads();
    }
}

// ---------------------------------------------------------------------------
// bf16 WMMA GEMM: C[rows x 256] = A[rows x 256] * B[256 x 256] + bias
// A is row-major bf16, B is supplied TRANSPOSED (BT[dout][din], row-major bf16).
// One wave32 per 16x16 C tile; K = 256 -> 8 x v_wmma_f32_16x16x32_bf16.
// Fragment layouts used here (contiguous k, same order in A and B):
//   A lane l: row = l%16; elems 0..15 -> K0 + e + (l<16 ? 0 : 16)
//   B lane l: col = l%16; elems 0..15 -> K0 + e + (l<16 ? 0 : 16)  (from BT row)
//   D lane l, vgpr r: (row0 + r + (l<16?0:8), col0 + l%16)
// ---------------------------------------------------------------------------
__global__ __launch_bounds__(32)
void gemm_bf16_wmma(const __bf16* __restrict__ A, const __bf16* __restrict__ BT,
                    const float* __restrict__ bias, float* __restrict__ C) {
    const int lane = threadIdx.x;
    const int row0 = blockIdx.y * 16;
    const int col0 = blockIdx.x * 16;
    const int hi   = (lane < 16) ? 0 : 1;
    const int l16  = lane & 15;

    const __bf16* arow = A  + (size_t)(row0 + l16) * DD;
    const __bf16* brow = BT + (size_t)(col0 + l16) * DD;

    v8f acc = {};
#pragma unroll
    for (int kk = 0; kk < 8; ++kk) {
        const int K0 = kk * 32;
        v8bf a0 = *(const v8bf*)(arow + K0 + hi * 16);       // K0+{0..7} or {16..23}
        v8bf a1 = *(const v8bf*)(arow + K0 + hi * 16 + 8);   // +8
        v16bf a;
#pragma unroll
        for (int e = 0; e < 8; ++e) { a[e] = a0[e]; a[e + 8] = a1[e]; }
        v16bf bmat = *(const v16bf*)(brow + K0 + hi * 16);   // 16 contiguous K
        acc = __builtin_amdgcn_wmma_f32_16x16x32_bf16(
                  false, a, false, bmat, (short)0, acc, false, false);
    }

    const float bsum = bias[col0 + l16];
#pragma unroll
    for (int r = 0; r < 8; ++r) {
        const int row = row0 + r + hi * 8;
        C[(size_t)row * DD + col0 + l16] = acc[r] + bsum;
    }
}

// ---------------------------------------------------------------------------
// Zero-fill float4 region (avg_attn block of d_out)
// ---------------------------------------------------------------------------
__global__ __launch_bounds__(256)
void zero_f4_kernel(float4* __restrict__ p, int n4) {
    int i = blockIdx.x * 256 + threadIdx.x;
    if (i < n4) p[i] = make_float4(0.f, 0.f, 0.f, 0.f);
}

// ---------------------------------------------------------------------------
// Sparse kNN attention: one wave32 per query row.
// Lanes 0..15 own the 16 neighbors (score + softmax); all 32 lanes do the
// weighted V gather (one d-channel each per head). Residual fused; writes
// x in bf16 for the Wo1 GEMM and scatters avg_attn.
// ---------------------------------------------------------------------------
__global__ __launch_bounds__(256)
void attn_kernel(const float* __restrict__ Q, const float* __restrict__ Km,
                 const float* __restrict__ Vm, const int* __restrict__ knn_idx,
                 const float* __restrict__ src_fea,
                 __bf16* __restrict__ x_bf, float* __restrict__ attn_out) {
    const int wave = threadIdx.x >> 5;
    const int lane = threadIdx.x & 31;
    const int row  = blockIdx.x * 8 + wave;   // 0..BN-1
    const int b    = row / NN;
    const float scale = 0.17677669529663687f; // 1/sqrt(32)

    int idx = 0;
    if (lane < KNNB) idx = knn_idx[(size_t)row * KNNB + lane];

    float attnsum = 0.f;
#pragma unroll 1
    for (int h = 0; h < HH; ++h) {
        const float* q = Q + (size_t)row * DD + h * DKK;
        float s = -3.4e38f;
        if (lane < KNNB) {
            const float* kp = Km + ((size_t)b * MM + idx) * DD + h * DKK;
            float acc = 0.f;
#pragma unroll
            for (int d = 0; d < DKK; ++d) acc += q[d] * kp[d];
            s = acc * scale;
        }
        // wave max (upper lanes hold -inf)
        float mx = s;
        for (int o = 16; o > 0; o >>= 1) mx = fmaxf(mx, __shfl_xor(mx, o, 32));
        float e = (lane < KNNB) ? __expf(s - mx) : 0.f;
        float sum = e;
        for (int o = 16; o > 0; o >>= 1) sum += __shfl_xor(sum, o, 32);
        const float a = e / sum;
        if (lane < KNNB) attnsum += a;

        // out[d] = sum_j a_j * V[idx_j][h*32 + d], d = lane
        float acc = 0.f;
#pragma unroll
        for (int j = 0; j < KNNB; ++j) {
            const float aj = __shfl(a, j, 32);
            const int   ij = __shfl(idx, j, 32);
            acc += aj * Vm[((size_t)b * MM + ij) * DD + h * DKK + lane];
        }
        const float xv = acc + src_fea[(size_t)row * DD + h * DKK + lane];
        x_bf[(size_t)row * DD + h * DKK + lane] = (__bf16)xv;
    }

    if (lane < KNNB)
        attn_out[(size_t)row * MM + idx] = attnsum * 0.125f;   // mean over 8 heads
}

// ---------------------------------------------------------------------------
// LayerNorm + ReLU over rows of y [BN x 256] -> h (bf16)
// ---------------------------------------------------------------------------
__global__ __launch_bounds__(256)
void ln_relu_kernel(const float* __restrict__ y, const float* __restrict__ g,
                    const float* __restrict__ bln, __bf16* __restrict__ h_bf) {
    __shared__ float red[256];
    const int row = blockIdx.x;
    const int tid = threadIdx.x;
    const float v = y[(size_t)row * DD + tid];

    red[tid] = v; __syncthreads();
    for (int s = 128; s > 0; s >>= 1) { if (tid < s) red[tid] += red[tid + s]; __syncthreads(); }
    const float mu = red[0] * (1.0f / DD);
    __syncthreads();

    const float dv = v - mu;
    red[tid] = dv * dv; __syncthreads();
    for (int s = 128; s > 0; s >>= 1) { if (tid < s) red[tid] += red[tid + s]; __syncthreads(); }
    const float var = red[0] * (1.0f / DD);

    float hv = dv * rsqrtf(var + 1e-5f) * g[tid] + bln[tid];
    hv = fmaxf(hv, 0.f);
    h_bf[(size_t)row * DD + tid] = (__bf16)hv;
}

// ---------------------------------------------------------------------------
// Launch
// ---------------------------------------------------------------------------
static inline size_t alignup(size_t x) { return (x + 255) & ~(size_t)255; }

extern "C" void kernel_launch(void* const* d_in, const int* in_sizes, int n_in,
                              void* d_out, int out_size, void* d_ws, size_t ws_size,
                              hipStream_t stream) {
    (void)in_sizes; (void)n_in; (void)out_size; (void)ws_size;

    const float* src     = (const float*)d_in[0];
    const float* tgt     = (const float*)d_in[1];
    const float* src_fea = (const float*)d_in[2];
    const float* tgt_fea = (const float*)d_in[3];
    const float* Wq  = (const float*)d_in[4];
    const float* bq  = (const float*)d_in[5];
    const float* Wk  = (const float*)d_in[6];
    const float* bk  = (const float*)d_in[7];
    const float* Wv  = (const float*)d_in[8];
    const float* bv  = (const float*)d_in[9];
    const float* Wo1 = (const float*)d_in[10];
    const float* bo1 = (const float*)d_in[11];
    const float* lng = (const float*)d_in[12];
    const float* lnb = (const float*)d_in[13];
    const float* Wo2 = (const float*)d_in[14];
    const float* bo2 = (const float*)d_in[15];

    float* out_upd  = (float*)d_out;                 // [BN, 256]
    float* out_attn = (float*)d_out + (size_t)BN * DD;  // [BN, MM]

    // Workspace carve-up
    char* w = (char*)d_ws;
    int*    knn = (int*)w;        w += alignup((size_t)BN * KNNB * sizeof(int));
    float*  Qf  = (float*)w;      w += alignup((size_t)BN * DD * sizeof(float));
    float*  Kf  = (float*)w;      w += alignup((size_t)BM * DD * sizeof(float));
    float*  Vf  = (float*)w;      w += alignup((size_t)BM * DD * sizeof(float));
    float*  Yf  = (float*)w;      w += alignup((size_t)BN * DD * sizeof(float));
    __bf16* sfb = (__bf16*)w;     w += alignup((size_t)BN * DD * sizeof(__bf16));
    __bf16* tfb = (__bf16*)w;     w += alignup((size_t)BM * DD * sizeof(__bf16));
    __bf16* xb  = (__bf16*)w;     w += alignup((size_t)BN * DD * sizeof(__bf16));
    __bf16* hb  = (__bf16*)w;     w += alignup((size_t)BN * DD * sizeof(__bf16));
    __bf16* wqT  = (__bf16*)w;    w += alignup((size_t)DD * DD * sizeof(__bf16));
    __bf16* wkT  = (__bf16*)w;    w += alignup((size_t)DD * DD * sizeof(__bf16));
    __bf16* wvT  = (__bf16*)w;    w += alignup((size_t)DD * DD * sizeof(__bf16));
    __bf16* wo1T = (__bf16*)w;    w += alignup((size_t)DD * DD * sizeof(__bf16));
    __bf16* wo2T = (__bf16*)w;    w += alignup((size_t)DD * DD * sizeof(__bf16));

    // 1) bf16 conversions + weight transposes
    cvt_bf16_kernel<<<(BN * DD + 255) / 256, 256, 0, stream>>>(src_fea, sfb, BN * DD);
    cvt_bf16_kernel<<<(BM * DD + 255) / 256, 256, 0, stream>>>(tgt_fea, tfb, BM * DD);
    transpose_cvt_kernel<<<(DD * DD) / 256, 256, 0, stream>>>(Wq,  wqT);
    transpose_cvt_kernel<<<(DD * DD) / 256, 256, 0, stream>>>(Wk,  wkT);
    transpose_cvt_kernel<<<(DD * DD) / 256, 256, 0, stream>>>(Wv,  wvT);
    transpose_cvt_kernel<<<(DD * DD) / 256, 256, 0, stream>>>(Wo1, wo1T);
    transpose_cvt_kernel<<<(DD * DD) / 256, 256, 0, stream>>>(Wo2, wo2T);

    // 2) kNN selection
    knn_kernel<<<BN, 256, 0, stream>>>(src, tgt, knn);

    // 3) Q/K/V projections (bf16 WMMA)
    gemm_bf16_wmma<<<dim3(DD / 16, BN / 16), 32, 0, stream>>>(sfb, wqT, bq, Qf);
    gemm_bf16_wmma<<<dim3(DD / 16, BM / 16), 32, 0, stream>>>(tfb, wkT, bk, Kf);
    gemm_bf16_wmma<<<dim3(DD / 16, BM / 16), 32, 0, stream>>>(tfb, wvT, bv, Vf);

    // 4) zero avg_attn then sparse attention (+ residual, x -> bf16)
    zero_f4_kernel<<<((size_t)BN * MM / 4 + 255) / 256, 256, 0, stream>>>(
        (float4*)out_attn, (int)((size_t)BN * MM / 4));
    attn_kernel<<<BN / 8, 256, 0, stream>>>(Qf, Kf, Vf, knn, src_fea, xb, out_attn);

    // 5) y = x @ Wo1 + bo1 ; LN + ReLU ; updated = h @ Wo2 + bo2
    gemm_bf16_wmma<<<dim3(DD / 16, BN / 16), 32, 0, stream>>>(xb, wo1T, bo1, Yf);
    ln_relu_kernel<<<BN, 256, 0, stream>>>(Yf, lng, lnb, hb);
    gemm_bf16_wmma<<<dim3(DD / 16, BN / 16), 32, 0, stream>>>(hb, wo2T, bo2, out_upd);
}
